// McMotLoss_68229850464577
// MI455X (gfx1250) — matrix operand order repaired and checked by reading, not compiled, not run on hardware
//
#include <hip/hip_runtime.h>
#include <hip/hip_bf16.h>

// ---------------- problem constants ----------------
#define Bc    4
#define NCc   5
#define Hc    152
#define Wc    272
#define Dc    128
#define NIDc  500
#define KOBJ  128
#define KID   256
#define HWc   (Hc * Wc)      // 41344
#define Ntot  (Bc * HWc)     // 165376
#define NPAD  512            // 2*K_ID rows; also padded NID for logits cols
#define EMB_SCALE 8.7859508f // sqrt(2)*ln(499)

typedef __attribute__((ext_vector_type(16))) _Float16 v16h;
typedef __attribute__((ext_vector_type(8)))  float    v8f;

// ---------------- scratch scalar slots ----------------
// [0..2]  hm: pos_loss, neg_loss, num_pos
// [3..5]  pre_hm: pos_loss, neg_loss, num_pos
// [8..11] wh, pre_wh, off, pre_off (final ratios)
// [16]    ce_sum (all classes)   [17] trip_sum (all classes)

__global__ void mcmot_init_kernel(float* s) {
  if (threadIdx.x < 32) s[threadIdx.x] = 0.0f;
}

// ---------------- focal loss reduction ----------------
__global__ void mcmot_focal_kernel(const float* __restrict__ x,
                                   const float* __restrict__ gt,
                                   float* acc3) {
  __shared__ float r0[256], r1[256], r2[256];
  const int NTOT = Bc * NCc * HWc;
  int tid = threadIdx.x;
  float pl = 0.f, nl = 0.f, np = 0.f;
  for (int i = blockIdx.x * blockDim.x + tid; i < NTOT; i += gridDim.x * blockDim.x) {
    __builtin_prefetch(&x[i + (gridDim.x * blockDim.x)], 0, 1);   // global_prefetch_b8
    float p = 1.f / (1.f + expf(-x[i]));
    p = fminf(fmaxf(p, 1e-4f), 1.f - 1e-4f);
    float g = gt[i];
    if (g == 1.f) {
      float q = 1.f - p;
      pl += logf(p) * q * q;
      np += 1.f;
    } else {
      float gg = 1.f - g;
      float g4 = gg * gg; g4 *= g4;
      nl += logf(1.f - p) * p * p * g4;
    }
  }
  r0[tid] = pl; r1[tid] = nl; r2[tid] = np;
  __syncthreads();
  for (int s = 128; s > 0; s >>= 1) {
    if (tid < s) { r0[tid] += r0[tid + s]; r1[tid] += r1[tid + s]; r2[tid] += r2[tid + s]; }
    __syncthreads();
  }
  if (tid == 0) {
    atomicAdd(&acc3[0], r0[0]);
    atomicAdd(&acc3[1], r1[0]);
    atomicAdd(&acc3[2], r2[0]);
  }
}

// ---------------- masked L1 (reg_l1), one block ----------------
__global__ void mcmot_reg_l1_kernel(const float* __restrict__ out2,   // (B,2,HW)
                                    const float* __restrict__ mask,   // (B,KOBJ)
                                    const int*   __restrict__ ind,    // (B,KOBJ)
                                    const float* __restrict__ tgt,    // (B,KOBJ,2)
                                    float* slot) {
  __shared__ float rn[256], rm[256];
  int t = threadIdx.x;
  float num = 0.f, msum = 0.f;
  for (int i = t; i < Bc * KOBJ; i += 256) {
    int b = i / KOBJ;
    float m = mask[i];
    int pos = ind[i];
#pragma unroll
    for (int ch = 0; ch < 2; ++ch) {
      float pred = out2[(b * 2 + ch) * HWc + pos];
      float tg = tgt[i * 2 + ch];
      num += fabsf(pred * m - tg * m);
    }
    msum += m;
  }
  rn[t] = num; rm[t] = msum;
  __syncthreads();
  for (int s = 128; s > 0; s >>= 1) {
    if (t < s) { rn[t] += rn[t + s]; rm[t] += rm[t + s]; }
    __syncthreads();
  }
  if (t == 0) *slot = rn[0] / (rm[0] * 2.f + 1e-4f);
}

// ---------------- top_k(==c) index selection (ascending, exact 256 matches) ----------------
__global__ void mcmot_select_kernel(const int* __restrict__ cmap,
                                    const int* __restrict__ pmap,
                                    const int* __restrict__ tr,      // (B,NC,HW)
                                    const int* __restrict__ pre_tr,  // (B,NC,HW)
                                    int c, int* idx_buf, int* tgt_buf) {
  int t = threadIdx.x;                     // 0..511
  const int* map = (t < KID) ? cmap : pmap;
  int want = (t < KID) ? t : t - KID;
  int found = -1, cnt = 0;
  for (int n = 0; n < Ntot; ++n) {
    if (map[n] == c) {
      if (cnt == want) { found = n; break; }
      ++cnt;
    }
  }
  int n = (found >= 0) ? found : 0;
  idx_buf[t] = n;
  int b = n / HWc, hw = n % HWc;
  const int* trs = (t < KID) ? tr : pre_tr;
  tgt_buf[t] = trs[(b * NCc + c) * HWc + hw];
}

// ---------------- gather rows, l2-normalize, scale, store f16; sq from f16 ----------------
__global__ void mcmot_gather_norm_kernel(const float* __restrict__ id_out,
                                         const float* __restrict__ pre_id,
                                         const int* __restrict__ idx_buf,
                                         _Float16* headh, float* sq) {
  __shared__ float red[128];
  int j = blockIdx.x;     // 0..511
  int d = threadIdx.x;    // 0..127
  int n = idx_buf[j];
  const float* src = (j < KID) ? id_out : pre_id;
  int b = n / HWc, hw = n % HWc;
  float x = src[(b * Dc + d) * HWc + hw];
  red[d] = x * x;
  __syncthreads();
  for (int s = 64; s > 0; s >>= 1) {
    if (d < s) red[d] += red[d + s];
    __syncthreads();
  }
  float nrm = sqrtf(red[0]);
  __syncthreads();
  float v = EMB_SCALE * x / fmaxf(nrm, 1e-12f);
  _Float16 h = (_Float16)v;
  headh[j * Dc + d] = h;
  float v2 = (float)h;
  red[d] = v2 * v2;
  __syncthreads();
  for (int s = 64; s > 0; s >>= 1) {
    if (d < s) red[d] += red[d + s];
    __syncthreads();
  }
  if (d == 0) sq[j] = red[0];
}

// ---------------- f16-pad cls_w[c] to 512x128 ----------------
__global__ void mcmot_wpad_kernel(const float* __restrict__ cls_w, int c, _Float16* wpad) {
  int i = blockIdx.x * blockDim.x + threadIdx.x;    // 0 .. 512*128-1
  int r = i / Dc, d = i % Dc;
  float v = (r < NIDc) ? cls_w[(c * NIDc + r) * Dc + d] : 0.f;
  wpad[i] = (_Float16)v;
}

// ---------------- WMMA fragment loaders (contiguous k = 16h + s order used by A and B alike, wave32) ----------------
// A 16x32 f16: lanes 0-15 -> M=lane, K=0..15; lanes 16-31 -> M=lane-16, K=16..31
__device__ inline v16h mcmot_load_A(const _Float16* M, int rowBase, int kOff, int lane) {
  int m = lane & 15;
  int kb = (lane >> 4) * 16;
  const _Float16* row = M + (rowBase + m) * Dc + kOff;
  v16h a;
#pragma unroll
  for (int hh = 0; hh < 16; ++hh) {
    int K = kb + hh;
    a[hh] = row[K];
  }
  return a;
}
// B 32x16 f16 where B[k][n] = Rows[colBase+n][k] (i.e. multiply by Rows^T):
// lanes 0-15 -> N=lane, K=kOff+0..15; lanes 16-31 -> N=lane-16, K=kOff+16..31
__device__ inline v16h mcmot_load_BT(const _Float16* Rows, int colBase, int kOff, int lane) {
  int n = lane & 15;
  int kb = (lane >> 4) * 16;
  const _Float16* row = Rows + (colBase + n) * Dc + kOff + kb;
  v16h b;
#pragma unroll
  for (int hh = 0; hh < 16; ++hh) b[hh] = row[hh];
  return b;
}

// ---------------- logits = head @ wpad^T + bias (512x512, cols>=500 -> -1e30) ----------------
__global__ void mcmot_logits_wmma_kernel(const _Float16* __restrict__ headh,
                                         const _Float16* __restrict__ wpad,
                                         const float* __restrict__ cls_b,
                                         int c, float* __restrict__ logits) {
  int lane = threadIdx.x;               // one wave32 per 16x16 tile
  int tm = blockIdx.x >> 5, tn = blockIdx.x & 31;
  v8f acc = {};
#pragma unroll
  for (int kk = 0; kk < Dc; kk += 32) {
    v16h a = mcmot_load_A(headh, tm * 16, kk, lane);
    v16h b = mcmot_load_BT(wpad, tn * 16, kk, lane);
    acc = __builtin_amdgcn_wmma_f32_16x16x32_f16(false, a, false, b, (short)0, acc, false, false);
  }
  int n = lane & 15;
  int col = tn * 16 + n;
#pragma unroll
  for (int r = 0; r < 8; ++r) {
    int row = tm * 16 + (lane >> 4) * 8 + r;
    logits[row * NPAD + col] = (col < NIDc) ? (acc[r] + cls_b[c * NIDc + col]) : -1e30f;
  }
}

// ---------------- gram = head @ head^T (512x512) ----------------
__global__ void mcmot_gram_wmma_kernel(const _Float16* __restrict__ headh,
                                       float* __restrict__ gram) {
  int lane = threadIdx.x;
  int tm = blockIdx.x >> 5, tn = blockIdx.x & 31;
  v8f acc = {};
#pragma unroll
  for (int kk = 0; kk < Dc; kk += 32) {
    v16h a = mcmot_load_A(headh, tm * 16, kk, lane);
    v16h b = mcmot_load_BT(headh, tn * 16, kk, lane);
    acc = __builtin_amdgcn_wmma_f32_16x16x32_f16(false, a, false, b, (short)0, acc, false, false);
  }
  int n = lane & 15;
  int col = tn * 16 + n;
#pragma unroll
  for (int r = 0; r < 8; ++r) {
    int row = tm * 16 + (lane >> 4) * 8 + r;
    gram[row * NPAD + col] = acc[r];
  }
}

// ---------------- cross-entropy mean contribution (accumulate sum of per-row CE) ----------------
__global__ void mcmot_ce_kernel(const float* __restrict__ logits,
                                const int* __restrict__ tgt_buf,
                                float* ce_acc) {
  __shared__ float red[128];
  int i = blockIdx.x, t = threadIdx.x;
  const float* row = logits + i * NPAD;
  float mx = -1e30f;
  for (int j = t; j < NPAD; j += 128) mx = fmaxf(mx, row[j]);
  red[t] = mx;
  __syncthreads();
  for (int s = 64; s > 0; s >>= 1) {
    if (t < s) red[t] = fmaxf(red[t], red[t + s]);
    __syncthreads();
  }
  mx = red[0];
  __syncthreads();
  float sm = 0.f;
  for (int j = t; j < NIDc; j += 128) sm += expf(row[j] - mx);
  red[t] = sm;
  __syncthreads();
  for (int s = 64; s > 0; s >>= 1) {
    if (t < s) red[t] += red[t + s];
    __syncthreads();
  }
  if (t == 0) {
    float lse = mx + logf(red[0]);
    int lbl = tgt_buf[i];
    atomicAdd(ce_acc, -(row[lbl] - lse));
  }
}

// ---------------- triplet (accumulate sum of per-row hinge) ----------------
__global__ void mcmot_triplet_kernel(const float* __restrict__ gram,
                                     const float* __restrict__ sq,
                                     const int* __restrict__ tgt_buf,
                                     float* trip_acc) {
  __shared__ float rmax[128], rmin[128];
  int i = blockIdx.x, t = threadIdx.x;
  int li = tgt_buf[i];
  float sqi = sq[i];
  float ap = -1e30f, an = 1e30f;
  for (int j = t; j < NPAD; j += 128) {
    float d2 = sqi + sq[j] - 2.f * gram[i * NPAD + j];
    float d = sqrtf(fmaxf(d2, 1e-12f));
    if (tgt_buf[j] == li) ap = fmaxf(ap, d);
    else an = fminf(an, d);
  }
  rmax[t] = ap; rmin[t] = an;
  __syncthreads();
  for (int s = 64; s > 0; s >>= 1) {
    if (t < s) { rmax[t] = fmaxf(rmax[t], rmax[t + s]); rmin[t] = fminf(rmin[t], rmin[t + s]); }
    __syncthreads();
  }
  if (t == 0) atomicAdd(trip_acc, fmaxf(rmax[0] - rmin[0] + 0.3f, 0.f));
}

// ---------------- final scalar combination ----------------
__global__ void mcmot_finalize_kernel(const float* __restrict__ s,
                                      const float* __restrict__ s_det,
                                      const float* __restrict__ s_id,
                                      float* out) {
  float hm   = (s[2] > 0.f) ? -(s[0] + s[1]) / fmaxf(s[2], 1.f) : -s[1];
  float phm  = (s[5] > 0.f) ? -(s[3] + s[4]) / fmaxf(s[5], 1.f) : -s[4];
  float det  = hm  + 0.1f * s[8] + s[10];
  float pdet = phm + 0.1f * s[9] + s[11];
  // reid = sum_c (ce_mean_c + trip_mean_c) / (2*K_ID); means over 512 rows
  float reid = (s[16] + s[17]) / (512.f * 512.f);
  float sd = s_det[0], si = s_id[0];
  float loss  = expf(-sd) * det  + expf(-si) * reid + (sd + si);
  float ploss = expf(-sd) * pdet + (sd + si);
  out[0] = 0.5f * (loss + ploss);
}

extern "C" void kernel_launch(void* const* d_in, const int* in_sizes, int n_in,
                              void* d_out, int out_size, void* d_ws, size_t ws_size,
                              hipStream_t stream) {
  (void)in_sizes; (void)n_in; (void)out_size; (void)ws_size;
  const float* hm_out      = (const float*)d_in[0];
  const float* wh_out      = (const float*)d_in[1];
  const float* reg_out     = (const float*)d_in[2];
  const float* id_out      = (const float*)d_in[3];
  const float* pre_hm_out  = (const float*)d_in[4];
  const float* pre_wh_out  = (const float*)d_in[5];
  const float* pre_reg_out = (const float*)d_in[6];
  const float* pre_id_out  = (const float*)d_in[7];
  const float* hm_gt       = (const float*)d_in[8];
  const float* pre_hm_gt   = (const float*)d_in[9];
  const float* wh_gt       = (const float*)d_in[10];
  const float* pre_wh_gt   = (const float*)d_in[11];
  const float* reg_gt      = (const float*)d_in[12];
  const float* pre_reg_gt  = (const float*)d_in[13];
  const float* reg_mask    = (const float*)d_in[14];
  const float* pre_reg_mask= (const float*)d_in[15];
  const float* cls_w       = (const float*)d_in[16];
  const float* cls_b       = (const float*)d_in[17];
  const float* s_det       = (const float*)d_in[18];
  const float* s_id        = (const float*)d_in[19];
  const int*   ind         = (const int*)d_in[20];
  const int*   pre_ind     = (const int*)d_in[21];
  const int*   cls_id_map  = (const int*)d_in[22];
  const int*   pre_cls_id_map = (const int*)d_in[23];
  const int*   cls_tr_ids  = (const int*)d_in[24];
  const int*   pre_cls_tr_ids = (const int*)d_in[25];
  float* out = (float*)d_out;

  // workspace layout
  char* ws = (char*)d_ws;
  float*     sc      = (float*)ws;                         //  32 f32 scalar slots
  int*       idx_buf = (int*)(ws + 256);                   // 512 ints
  int*       tgt_buf = (int*)(ws + 256 + 2048);            // 512 ints
  float*     sq      = (float*)(ws + 256 + 4096);          // 512 f32
  _Float16*  headh   = (_Float16*)(ws + 8192);             // 512x128 f16  (128 KB)
  _Float16*  wpad    = (_Float16*)(ws + 8192 + 131072);    // 512x128 f16  (128 KB)
  float*     logits  = (float*)(ws + 8192 + 262144);       // 512x512 f32  (1 MB)
  float*     gram    = (float*)(ws + 8192 + 262144 + 1048576); // 512x512 f32 (1 MB)

  mcmot_init_kernel<<<1, 64, 0, stream>>>(sc);

  // focal losses (3-way reductions)
  mcmot_focal_kernel<<<1024, 256, 0, stream>>>(hm_out,     hm_gt,     sc + 0);
  mcmot_focal_kernel<<<1024, 256, 0, stream>>>(pre_hm_out, pre_hm_gt, sc + 3);

  // masked L1 terms
  mcmot_reg_l1_kernel<<<1, 256, 0, stream>>>(wh_out,      reg_mask,     ind,     wh_gt,      sc + 8);
  mcmot_reg_l1_kernel<<<1, 256, 0, stream>>>(pre_wh_out,  pre_reg_mask, pre_ind, pre_wh_gt,  sc + 9);
  mcmot_reg_l1_kernel<<<1, 256, 0, stream>>>(reg_out,     reg_mask,     ind,     reg_gt,     sc + 10);
  mcmot_reg_l1_kernel<<<1, 256, 0, stream>>>(pre_reg_out, pre_reg_mask, pre_ind, pre_reg_gt, sc + 11);

  // per-class ReID loss (classes sequential; buffers reused)
  for (int c = 0; c < NCc; ++c) {
    mcmot_select_kernel<<<1, 512, 0, stream>>>(cls_id_map, pre_cls_id_map,
                                               cls_tr_ids, pre_cls_tr_ids,
                                               c, idx_buf, tgt_buf);
    mcmot_gather_norm_kernel<<<512, 128, 0, stream>>>(id_out, pre_id_out, idx_buf, headh, sq);
    mcmot_wpad_kernel<<<(NPAD * Dc) / 256, 256, 0, stream>>>(cls_w, c, wpad);
    mcmot_logits_wmma_kernel<<<32 * 32, 32, 0, stream>>>(headh, wpad, cls_b, c, logits);
    mcmot_gram_wmma_kernel<<<32 * 32, 32, 0, stream>>>(headh, gram);
    mcmot_ce_kernel<<<512, 128, 0, stream>>>(logits, tgt_buf, sc + 16);
    mcmot_triplet_kernel<<<512, 128, 0, stream>>>(gram, sq, tgt_buf, sc + 17);
  }

  mcmot_finalize_kernel<<<1, 1, 0, stream>>>(sc, s_det, s_id, out);
}
